// DCEModulatedResBlock_11347303596578
// MI455X (gfx1250) — compile-verified, matrix operand order checked by reading, not run on hardware
//
#include <hip/hip_runtime.h>
#include <hip/hip_bf16.h>

// ---------------------------------------------------------------------------
// DCE-modulated ResBlock for MI455X (gfx1250, wave32, WMMA).
// Shapes: B=16, C=128, H=W=128, L2=100.
// Heavy convs: implicit GEMMs on v_wmma_f32_16x16x32_bf16, f32 accumulate,
// BN batch statistics fused into the conv kernels. LDS staging uses
// GLOBAL_LOAD_ASYNC_TO_LDS_B128 when the toolchain exposes it.
// ---------------------------------------------------------------------------

typedef __attribute__((ext_vector_type(16))) __bf16 v16bf;
typedef __attribute__((ext_vector_type(8)))  float  v8f;
typedef __attribute__((ext_vector_type(4)))  int    v4i;

union AFrag { v16bf v; uint4 q[2]; };

#define NB   16
#define NC   128
#define NH   128
#define NW   128
#define HW   16384      // NH*NW
#define NPIX 262144     // NB*HW
#define KDCE 12800      // L2*C

#if defined(__HIP_DEVICE_COMPILE__) && __has_builtin(__builtin_amdgcn_global_load_async_to_lds_b128)
#define HAVE_ASYNC_LDS 1
typedef __attribute__((address_space(1))) v4i v4i_as1;   // prints as "__device__"
typedef __attribute__((address_space(3))) v4i v4i_as3;   // LDS
#define ASYNC_CP16(gp, lp)                                                    \
    __builtin_amdgcn_global_load_async_to_lds_b128(                           \
        (v4i_as1*)(const void*)(gp), (v4i_as3*)(void*)(lp), 0, 0)
#if __has_builtin(__builtin_amdgcn_s_wait_asynccnt)
#define ASYNC_WAIT() __builtin_amdgcn_s_wait_asynccnt(0)
#else
#define ASYNC_WAIT() asm volatile("s_wait_asynccnt 0x0" ::: "memory")
#endif
#endif

__device__ __forceinline__ float dev_gelu(float x) {
    return 0.5f * x * (1.0f + erff(x * 0.70710678118654752f));
}
__device__ __forceinline__ float dev_silu(float x) {
    return x / (1.0f + expf(-x));
}

// -------------------------------------------------------------------- init
__global__ void k_init(float* stats) {
    for (int i = threadIdx.x; i < 768; i += 256) stats[i] = 0.0f;
}

// ---------------------------------------------------------- weight packing
// B-fragment lane layout (16-bit, 32x16): lane = half*16 + n;
//   e[j] -> K = (j<8 ? half*8 + j : 16 + half*8 + (j-8)), N = n.
__global__ void k_pack(const float* __restrict__ w1,   // [128,128,3,3]
                       const float* __restrict__ w2,   // [128,128,1,1]
                       const float* __restrict__ wsc,  // [128,128,1,1]
                       __bf16* __restrict__ f1, __bf16* __restrict__ f2,
                       __bf16* __restrict__ fsc) {
    int idx = blockIdx.x * 256 + threadIdx.x;
    if (idx < 147456) {                       // 36 slices * 8 ntiles * 32 * 16
        int j = idx & 15, lane = (idx >> 4) & 31, nt = (idx >> 9) & 7, s = idx >> 12;
        int half = lane >> 4, n = lane & 15;
        int k = (j < 8) ? (half * 8 + j) : (16 + half * 8 + (j - 8));
        int tap = s >> 2, cin = ((s & 3) << 5) + k, cout = nt * 16 + n;
        f1[idx] = (__bf16)w1[(cout * 128 + cin) * 9 + tap];
    } else if (idx < 147456 + 16384) {        // 4 slices * 8 * 32 * 16
        int t = idx - 147456;
        int j = t & 15, lane = (t >> 4) & 31, nt = (t >> 9) & 7, g = t >> 12;
        int half = lane >> 4, n = lane & 15;
        int k = (j < 8) ? (half * 8 + j) : (16 + half * 8 + (j - 8));
        f2[t] = (__bf16)w2[(nt * 16 + n) * 128 + ((g << 5) + k)];
    } else if (idx < 147456 + 32768) {
        int t = idx - 147456 - 16384;
        int j = t & 15, lane = (t >> 4) & 31, nt = (t >> 9) & 7, g = t >> 12;
        int half = lane >> 4, n = lane & 15;
        int k = (j < 8) ? (half * 8 + j) : (16 + half * 8 + (j - 8));
        fsc[t] = (__bf16)wsc[(nt * 16 + n) * 128 + ((g << 5) + k)];
    }
}

// ----------------------------------------------- depthwise 3x3 + avg pool
__global__ __launch_bounds__(256)
void k_spatial(const float* __restrict__ x, const float* __restrict__ dwc,
               float* __restrict__ spatial) {
    int bc = blockIdx.x;               // (b,c)
    int c = bc & 127;
    const float* xp = x + (size_t)bc * HW;
    float w[9];
#pragma unroll
    for (int i = 0; i < 9; ++i) w[i] = dwc[c * 9 + i];
    float acc = 0.0f;
    for (int p = threadIdx.x; p < HW; p += 256) {
        int hh = p >> 7, ww = p & 127;
        float s = 0.0f;
#pragma unroll
        for (int t = 0; t < 9; ++t) {
            int h2 = hh + t / 3 - 1, w2 = ww + t % 3 - 1;
            if (h2 >= 0 && h2 < NH && w2 >= 0 && w2 < NW)
                s += w[t] * xp[h2 * NW + w2];
        }
        acc += s;
    }
    __shared__ float red[256];
    red[threadIdx.x] = acc;
    __syncthreads();
    for (int o = 128; o > 0; o >>= 1) {
        if (threadIdx.x < o) red[threadIdx.x] += red[threadIdx.x + o];
        __syncthreads();
    }
    if (threadIdx.x == 0) spatial[bc] = red[0] * (1.0f / (float)HW);
}

// --------------------------------------------------- dce layer 1 (K=12800)
__global__ __launch_bounds__(256)
void k_dce1(const float* __restrict__ dce, const float* __restrict__ W1,
            const float* __restrict__ b1, float* __restrict__ h1) {
    int gw = (blockIdx.x * 256 + threadIdx.x) >> 5;   // one wave per output
    int lane = threadIdx.x & 31;
    int b = gw >> 7, c = gw & 127;
    float acc = 0.0f;
    for (int k = lane; k < KDCE; k += 32)
        acc += dce[b * KDCE + k] * W1[(size_t)k * 128 + c];
    for (int o = 16; o > 0; o >>= 1) acc += __shfl_xor(acc, o, 32);
    if (lane == 0) h1[gw] = dev_gelu(acc + b1[c]);
}

// ----------------------------------------------- SE chain -> mod_w [16,128]
__global__ void k_mod(const float* __restrict__ h1, const float* __restrict__ spatial,
                      const float* __restrict__ W2, const float* __restrict__ b2,
                      const float* __restrict__ Wsh, const float* __restrict__ bsh,
                      const float* __restrict__ Wex, const float* __restrict__ bex,
                      float* __restrict__ modw) {
    __shared__ float sm[16][128];
    __shared__ float ssh[16][64];
    int c = threadIdx.x;   // 128 threads
    for (int b = 0; b < NB; ++b) {
        float a = 0.0f;
        for (int k = 0; k < 128; ++k) a += h1[b * 128 + k] * W2[k * 128 + c];
        sm[b][c] = (a + b2[c]) * spatial[b * 128 + c];
    }
    __syncthreads();
    if (c < 64) {
        for (int b = 0; b < NB; ++b) {
            float a = 0.0f;
            for (int k = 0; k < 128; ++k) a += sm[b][k] * Wsh[k * 64 + c];
            ssh[b][c] = dev_gelu(a + bsh[c]);
        }
    }
    __syncthreads();
    for (int b = 0; b < NB; ++b) {
        float a = 0.0f;
        for (int k = 0; k < 64; ++k) a += ssh[b][k] * Wex[k * 128 + c];
        modw[b * 128 + c] = 1.0f / (1.0f + expf(-(a + bex[c])));
    }
}

// ---------------------- NCHW f32 -> NHWC bf16 transpose, fused * mod_w
__global__ __launch_bounds__(256)
void k_xm(const float* __restrict__ x, const float* __restrict__ modw,
          __bf16* __restrict__ xmT) {
    __shared__ float t[128][33];
    int tid = threadIdx.x;
    int b = blockIdx.y;
    size_t pix0 = (size_t)blockIdx.x * 32;
    for (int i = tid; i < 4096; i += 256) {          // coalesced NCHW read
        int c = i >> 5, p = i & 31;
        t[c][p] = x[((size_t)b * NC + c) * HW + pix0 + p] * modw[b * NC + c];
    }
    __syncthreads();
    for (int i = tid; i < 4096; i += 256) {          // coalesced NHWC write
        int p = i >> 7, c = i & 127;
        xmT[((size_t)b * HW + pix0 + p) * NC + c] = (__bf16)t[c][p];
    }
}

// -------------------- conv1 (3x3) + shortcut (1x1) via WMMA, fused BN sums
// Block: 32 pixels x 128 couts. 8 waves; each wave = 16 couts x 2 M-tiles.
__global__ __launch_bounds__(256)
void k_conv1sc(const __bf16* __restrict__ xmT, const __bf16* __restrict__ Wf1,
               const __bf16* __restrict__ Wfsc, float* __restrict__ y1,
               __bf16* __restrict__ ysc, float* __restrict__ stats) {
    __shared__ __align__(16) __bf16 tile[3][34][128];   // halo, 26112 B
    __shared__ float ssum[4][128];                      // sum1/sq1/sumsc/sqsc
    int tid = threadIdx.x;
    int wt = blockIdx.x;
    int b = wt >> 9, rem = wt & 511;
    int h = rem >> 2, w0 = (rem & 3) << 5;

    for (int i = tid; i < 512; i += 256) ((float*)ssum)[i] = 0.0f;

    // stage 3x34x128 bf16 halo (zero-padded), 16B granules
    const uint4* gsrc = (const uint4*)xmT;
    for (int q = tid; q < 1632; q += 256) {             // 102 pairs * 16 uint4
        int pair = q >> 4, wo = q & 15;
        int r = pair / 34, cc = pair % 34;
        int hh = h - 1 + r, ww = w0 - 1 + cc;
        if (hh >= 0 && hh < NH && ww >= 0 && ww < NW) {
            size_t pix = (size_t)b * HW + hh * NW + ww;
#ifdef HAVE_ASYNC_LDS
            ASYNC_CP16(gsrc + pix * 16 + wo, &((uint4*)tile)[pair * 16 + wo]);
#else
            ((uint4*)tile)[pair * 16 + wo] = gsrc[pix * 16 + wo];
#endif
        } else {
            uint4 z = {0u, 0u, 0u, 0u};
            ((uint4*)tile)[pair * 16 + wo] = z;
        }
    }
#ifdef HAVE_ASYNC_LDS
    ASYNC_WAIT();
#endif
    __syncthreads();

    int wave = tid >> 5, lane = tid & 31;
    int half = lane >> 4, lm = lane & 15;

    v8f acc1a = {0.f, 0.f, 0.f, 0.f, 0.f, 0.f, 0.f, 0.f};
    v8f acc1b = {0.f, 0.f, 0.f, 0.f, 0.f, 0.f, 0.f, 0.f};
    v8f accsa = {0.f, 0.f, 0.f, 0.f, 0.f, 0.f, 0.f, 0.f};
    v8f accsb = {0.f, 0.f, 0.f, 0.f, 0.f, 0.f, 0.f, 0.f};
    const uint4* wq1 = (const uint4*)Wf1;
    const uint4* wqs = (const uint4*)Wfsc;

    // conv1: K = 9 taps * 128 cin = 36 slices of 32
#pragma unroll 4
    for (int s = 0; s < 36; ++s) {
        int tap = s >> 2, cb = (s & 3) << 5;
        int dh = tap / 3, dw = tap % 3;
        AFrag bf, aa, ab;
        int boff = ((s * 8 + wave) * 32 + lane) * 2;
        bf.q[0] = wq1[boff];
        bf.q[1] = wq1[boff + 1];
        const __bf16* s0 = &tile[dh][lm + dw][cb + half * 8];
        aa.q[0] = *(const uint4*)s0;
        aa.q[1] = *(const uint4*)(s0 + 16);
        const __bf16* s1 = &tile[dh][lm + 16 + dw][cb + half * 8];
        ab.q[0] = *(const uint4*)s1;
        ab.q[1] = *(const uint4*)(s1 + 16);
        acc1a = __builtin_amdgcn_wmma_f32_16x16x32_bf16(
            false, aa.v, false, bf.v, (short)0, acc1a, false, false);
        acc1b = __builtin_amdgcn_wmma_f32_16x16x32_bf16(
            false, ab.v, false, bf.v, (short)0, acc1b, false, false);
    }
    // shortcut 1x1: K = 128 = 4 slices on the tile center row
#pragma unroll
    for (int g = 0; g < 4; ++g) {
        AFrag bf, aa, ab;
        int boff = ((g * 8 + wave) * 32 + lane) * 2;
        bf.q[0] = wqs[boff];
        bf.q[1] = wqs[boff + 1];
        const __bf16* s0 = &tile[1][lm + 1][(g << 5) + half * 8];
        aa.q[0] = *(const uint4*)s0;
        aa.q[1] = *(const uint4*)(s0 + 16);
        const __bf16* s1 = &tile[1][lm + 17][(g << 5) + half * 8];
        ab.q[0] = *(const uint4*)s1;
        ab.q[1] = *(const uint4*)(s1 + 16);
        accsa = __builtin_amdgcn_wmma_f32_16x16x32_bf16(
            false, aa.v, false, bf.v, (short)0, accsa, false, false);
        accsb = __builtin_amdgcn_wmma_f32_16x16x32_bf16(
            false, ab.v, false, bf.v, (short)0, accsb, false, false);
    }

    // D layout: lane holds column n = lane&15, rows m = half*8 + r
    size_t pixbase = (size_t)b * HW + h * NW + w0;
    int n = wave * 16 + lm;
    float s1 = 0, q1 = 0, ss = 0, qs = 0;
#pragma unroll
    for (int r = 0; r < 8; ++r) {
        int m = half * 8 + r;
        size_t ia = (pixbase + m) * NC + n;
        size_t ib = (pixbase + 16 + m) * NC + n;
        float va = acc1a[r], vb = acc1b[r];
        y1[ia] = va;  y1[ib] = vb;
        s1 += va + vb;  q1 += va * va + vb * vb;
        float wa = accsa[r], wb = accsb[r];
        ysc[ia] = (__bf16)wa;  ysc[ib] = (__bf16)wb;
        ss += wa + wb;  qs += wa * wa + wb * wb;
    }
    atomicAdd(&ssum[0][n], s1);
    atomicAdd(&ssum[1][n], q1);
    atomicAdd(&ssum[2][n], ss);
    atomicAdd(&ssum[3][n], qs);
    __syncthreads();
    if (tid < 128) {
        atomicAdd(&stats[tid],       ssum[0][tid]);
        atomicAdd(&stats[128 + tid], ssum[1][tid]);
        atomicAdd(&stats[256 + tid], ssum[2][tid]);
        atomicAdd(&stats[384 + tid], ssum[3][tid]);
    }
}

// ------------------------------------------- finalize BN1 + BNsc together
__global__ void k_bnstat1(const float* __restrict__ stats,
                          const float* __restrict__ g1, const float* __restrict__ b1,
                          const float* __restrict__ gsc, const float* __restrict__ bsc,
                          float* __restrict__ bnp) {
    int t = threadIdx.x;   // 256
    const float invN = 1.0f / (float)NPIX;
    if (t < 128) {
        float mean = stats[t] * invN;
        float var  = stats[128 + t] * invN - mean * mean;
        float sc   = g1[t] * rsqrtf(var + 1e-5f);
        bnp[t] = sc;  bnp[128 + t] = b1[t] - mean * sc;
    } else {
        int c = t - 128;
        float mean = stats[256 + c] * invN;
        float var  = stats[384 + c] * invN - mean * mean;
        float sc   = gsc[c] * rsqrtf(var + 1e-5f);
        bnp[256 + c] = sc;  bnp[384 + c] = bsc[c] - mean * sc;
    }
}

// ----------------- conv2 (1x1) on silu(bn1(y1)) via WMMA, fused BN2 sums
// Block: 32 pixels x 128 couts; raw y1 tile async-staged to LDS.
__global__ __launch_bounds__(256)
void k_conv2(const float* __restrict__ y1, const __bf16* __restrict__ Wf2,
             const float* __restrict__ bnp, __bf16* __restrict__ y2,
             float* __restrict__ stats2) {
#ifdef HAVE_ASYNC_LDS
    __shared__ __align__(16) float stage[32 * 128];     // raw y1 tile, 16 KB
#endif
    __shared__ __align__(16) __bf16 a2[32][128];        // activations, 8 KB
    __shared__ float ssum[2][128];
    int tid = threadIdx.x;
    size_t pixbase = (size_t)blockIdx.x * 32;

    ((float*)ssum)[tid] = 0.0f;
#ifdef HAVE_ASYNC_LDS
    {
        const uint4* src = (const uint4*)(y1 + pixbase * NC);   // 1024 uint4
        for (int i = tid; i < 1024; i += 256)
            ASYNC_CP16(src + i, &((uint4*)stage)[i]);
        ASYNC_WAIT();
        __syncthreads();
        for (int i = tid; i < 4096; i += 256) {
            int p = i >> 7, c = i & 127;
            float v = stage[i] * bnp[c] + bnp[128 + c];
            a2[p][c] = (__bf16)dev_silu(v);
        }
    }
#else
    for (int i = tid; i < 4096; i += 256) {
        int p = i >> 7, c = i & 127;
        float v = y1[(pixbase + p) * NC + c] * bnp[c] + bnp[128 + c];
        a2[p][c] = (__bf16)dev_silu(v);
    }
#endif
    __syncthreads();

    int wave = tid >> 5, lane = tid & 31;
    int half = lane >> 4, lm = lane & 15;
    v8f acca = {0.f, 0.f, 0.f, 0.f, 0.f, 0.f, 0.f, 0.f};
    v8f accb = {0.f, 0.f, 0.f, 0.f, 0.f, 0.f, 0.f, 0.f};
    const uint4* wq2 = (const uint4*)Wf2;
#pragma unroll
    for (int g = 0; g < 4; ++g) {
        AFrag bf, aa, ab;
        int boff = ((g * 8 + wave) * 32 + lane) * 2;
        bf.q[0] = wq2[boff];
        bf.q[1] = wq2[boff + 1];
        const __bf16* s0 = &a2[lm][(g << 5) + half * 8];
        aa.q[0] = *(const uint4*)s0;
        aa.q[1] = *(const uint4*)(s0 + 16);
        const __bf16* s1 = &a2[lm + 16][(g << 5) + half * 8];
        ab.q[0] = *(const uint4*)s1;
        ab.q[1] = *(const uint4*)(s1 + 16);
        acca = __builtin_amdgcn_wmma_f32_16x16x32_bf16(
            false, aa.v, false, bf.v, (short)0, acca, false, false);
        accb = __builtin_amdgcn_wmma_f32_16x16x32_bf16(
            false, ab.v, false, bf.v, (short)0, accb, false, false);
    }

    int n = wave * 16 + lm;
    float s2 = 0, q2 = 0;
#pragma unroll
    for (int r = 0; r < 8; ++r) {
        int m = half * 8 + r;
        float va = acca[r], vb = accb[r];
        y2[(pixbase + m) * NC + n]      = (__bf16)va;
        y2[(pixbase + 16 + m) * NC + n] = (__bf16)vb;
        s2 += va + vb;  q2 += va * va + vb * vb;
    }
    atomicAdd(&ssum[0][n], s2);
    atomicAdd(&ssum[1][n], q2);
    __syncthreads();
    if (tid < 128) {
        atomicAdd(&stats2[tid],       ssum[0][tid]);
        atomicAdd(&stats2[128 + tid], ssum[1][tid]);
    }
}

__global__ void k_bnstat2(const float* __restrict__ stats2,
                          const float* __restrict__ g2, const float* __restrict__ b2,
                          float* __restrict__ bnp2) {
    int c = threadIdx.x;   // 128
    const float invN = 1.0f / (float)NPIX;
    float mean = stats2[c] * invN;
    float var  = stats2[128 + c] * invN - mean * mean;
    float sc   = g2[c] * rsqrtf(var + 1e-5f);
    bnp2[c] = sc;  bnp2[128 + c] = b2[c] - mean * sc;
}

// ---------- final: silu(bn2(y2) + bnsc(ysc)), NHWC bf16 -> NCHW f32 out
__global__ __launch_bounds__(256)
void k_out(const __bf16* __restrict__ y2, const __bf16* __restrict__ ysc,
           const float* __restrict__ bnp2, const float* __restrict__ bnpsc,
           float* __restrict__ out) {
    __shared__ float t[32][129];
    int tid = threadIdx.x;
    int b = blockIdx.y;
    size_t pix0 = (size_t)blockIdx.x * 32;
    for (int i = tid; i < 4096; i += 256) {          // coalesced NHWC read
        int p = i >> 7, c = i & 127;
        size_t idx = ((size_t)b * HW + pix0 + p) * NC + c;
        float v2 = (float)y2[idx]  * bnp2[c]  + bnp2[128 + c];
        float vs = (float)ysc[idx] * bnpsc[c] + bnpsc[128 + c];
        t[p][c] = dev_silu(v2 + vs);
    }
    __syncthreads();
    for (int i = tid; i < 4096; i += 256) {          // coalesced NCHW write
        int c = i >> 5, p = i & 31;
        out[((size_t)b * NC + c) * HW + pix0 + p] = t[p][c];
    }
}

// ---------------------------------------------------------------------------
extern "C" void kernel_launch(void* const* d_in, const int* in_sizes, int n_in,
                              void* d_out, int out_size, void* d_ws, size_t ws_size,
                              hipStream_t stream) {
    (void)in_sizes; (void)n_in; (void)out_size; (void)ws_size;
    const float* x    = (const float*)d_in[0];
    const float* dce  = (const float*)d_in[1];
    const float* dwc  = (const float*)d_in[2];
    const float* W1   = (const float*)d_in[3];
    const float* b1   = (const float*)d_in[4];
    const float* W2   = (const float*)d_in[5];
    const float* b2   = (const float*)d_in[6];
    const float* Wsh  = (const float*)d_in[7];
    const float* bsh  = (const float*)d_in[8];
    const float* Wex  = (const float*)d_in[9];
    const float* bex  = (const float*)d_in[10];
    const float* c1w  = (const float*)d_in[11];
    const float* g1   = (const float*)d_in[12];
    const float* bb1  = (const float*)d_in[13];
    const float* c2w  = (const float*)d_in[14];
    const float* g2   = (const float*)d_in[15];
    const float* bb2  = (const float*)d_in[16];
    const float* scw  = (const float*)d_in[17];
    const float* gsc  = (const float*)d_in[18];
    const float* bsc  = (const float*)d_in[19];

    char* ws = (char*)d_ws;
    size_t off = 0;
    auto alloc = [&](size_t bytes) {
        size_t r = off;
        off = (off + bytes + 255) & ~(size_t)255;
        return r;
    };
    const size_t NE = (size_t)NPIX * NC;                  // 33.5M elems
    __bf16* xmT     = (__bf16*)(ws + alloc(NE * 2));      // NHWC bf16
    __bf16* ysc     = (__bf16*)(ws + alloc(NE * 2));      // shortcut raw, bf16
    __bf16* y2      = (__bf16*)(ws + alloc(NE * 2));      // conv2 raw, bf16
    float*  spatial = (float*)(ws + alloc(2048 * 4));
    float*  h1      = (float*)(ws + alloc(2048 * 4));
    float*  modw    = (float*)(ws + alloc(2048 * 4));
    float*  stats   = (float*)(ws + alloc(768 * 4));      // sum1,sq1,sumsc,sqsc,sum2,sq2
    float*  bnp     = (float*)(ws + alloc(768 * 4));      // scale/shift: bn1,bnsc,bn2
    __bf16* Wf1     = (__bf16*)(ws + alloc(147456 * 2));
    __bf16* Wf2     = (__bf16*)(ws + alloc(16384 * 2));
    __bf16* Wfsc    = (__bf16*)(ws + alloc(16384 * 2));
    float*  y1      = (float*)d_out;                      // conv1 raw f32 (scratch, NHWC)

    k_init   <<<1, 256, 0, stream>>>(stats);
    k_pack   <<<704, 256, 0, stream>>>(c1w, c2w, scw, Wf1, Wf2, Wfsc);
    k_spatial<<<2048, 256, 0, stream>>>(x, dwc, spatial);
    k_dce1   <<<64, 256, 0, stream>>>(dce, W1, b1, h1);
    k_mod    <<<1, 128, 0, stream>>>(h1, spatial, W2, b2, Wsh, bsh, Wex, bex, modw);
    k_xm     <<<dim3(512, 16), 256, 0, stream>>>(x, modw, xmT);
    k_conv1sc<<<8192, 256, 0, stream>>>(xmT, Wf1, Wfsc, y1, ysc, stats);
    k_bnstat1<<<1, 256, 0, stream>>>(stats, g1, bb1, gsc, bsc, bnp);
    k_conv2  <<<8192, 256, 0, stream>>>(y1, Wf2, bnp, y2, stats + 512);
    k_bnstat2<<<1, 128, 0, stream>>>(stats + 512, g2, bb2, bnp + 512);
    k_out    <<<dim3(512, 16), 256, 0, stream>>>(y2, ysc, bnp + 512, bnp + 256,
                                                 (float*)d_out);
}
